// Clusteror_35485019800225
// MI455X (gfx1250) — matrix operand order repaired and checked by reading, not compiled, not run on hardware
//
#include <hip/hip_runtime.h>
#include <hip/hip_bf16.h>
#include <stdint.h>

#define NTOT 200010   // N + P
#define NND  200000   // N
#define PV   10       // P

typedef __attribute__((ext_vector_type(16))) __bf16       v16bf;
typedef __attribute__((ext_vector_type(8)))  float        v8f;
typedef __attribute__((ext_vector_type(4)))  unsigned int v4u;

union Frag16 {
    v16bf          v;
    unsigned short u[16];
    v4u            q[2];
};

// One dynamic-LDS blob shared by all kernels (per-kernel sizes set at launch).
extern __shared__ char smem_raw[];

__device__ __forceinline__ unsigned short f2bf(float f) {
    __bf16 h = (__bf16)f;                       // native v_cvt_pk_bf16_f32 (RNE)
    return __builtin_bit_cast(unsigned short, h);
}

__device__ __forceinline__ v8f wmma_bf16(v16bf a, v16bf b, v8f c) {
    // D = A(16x32 bf16) * B(32x16 bf16) + C(16x16 f32)
    return __builtin_amdgcn_wmma_f32_16x16x32_bf16(
        /*neg_a=*/false, a, /*neg_b=*/false, b,
        /*c_mod=*/(short)0, c, /*reuse_a=*/false, /*reuse_b=*/false);
}

__device__ __forceinline__ Frag16 load_bfrag(const unsigned short* __restrict__ p,
                                             int frag, int lane) {
    Frag16 f;
    const v4u* q = (const v4u*)(p + ((size_t)frag << 9) + (lane << 4));
    f.q[0] = q[0];
    f.q[1] = q[1];
    return f;
}

__device__ __forceinline__ Frag16 load_afrag16(const unsigned short* p, int off0) {
    Frag16 a;
    a.q[0] = *(const v4u*)(p + off0);
    a.q[1] = *(const v4u*)(p + 16 + off0);
    return a;
}

__device__ __forceinline__ float elu(float v) {
    return v > 0.0f ? v : (__expf(v) - 1.0f);
}

// ---------------------------------------------------------------------------
// Pack a row-major f32 weight (Kdim x Ndim) into bf16 WMMA B-fragments.
// Fragment (kt,nt): lane L holds column n = nt*16 + (L&15),
//   K values kbase..kbase+7 then kbase+16..kbase+23 with kbase = kt*32 + (L>=16 ? 8 : 0).
// ---------------------------------------------------------------------------
__global__ __launch_bounds__(128) void pack_kernel(const float* __restrict__ src,
                                                   unsigned short* __restrict__ dst,
                                                   int Kdim, int Ndim) {
    const int wave = threadIdx.x >> 5;
    const int lane = threadIdx.x & 31;
    const int frag = blockIdx.x * (blockDim.x >> 5) + wave;
    const int ntl  = Ndim >> 4;
    const int nfrag = (Kdim >> 5) * ntl;
    if (frag >= nfrag) return;
    const int kt = frag / ntl;
    const int nt = frag - kt * ntl;
    const int n = nt * 16 + (lane & 15);
    const int kbase = kt * 32 + (lane >> 4) * 8;
    Frag16 f;
#pragma unroll
    for (int j = 0; j < 16; ++j)
        f.u[j] = f2bf(src[(size_t)(kbase + j + (j & 8)) * Ndim + n]);
    v4u* o = (v4u*)(dst + ((size_t)frag << 9) + (lane << 4));
    o[0] = f.q[0];
    o[1] = f.q[1];
}

// ---------------------------------------------------------------------------
// Layer 1: h1 = elu(LN(x @ W1 + b)) ; rows >= N read vnode_embed instead of x
// and get +vnode_bias_hid after ELU.  Output bf16 row-major (NTOT x 256).
// Persistent blocks; packed W1 (128 KB) staged in LDS once per block.
// No manual B chaining here: f32 A staging + cvts already saturate the
// register budget (round-4 chaining spilled); 4 waves/SIMD hide LDS latency.
// ---------------------------------------------------------------------------
__global__ __launch_bounds__(256)
__attribute__((amdgpu_waves_per_eu(4)))
void layer1_kernel(
    const float* __restrict__ x, const float* __restrict__ vemb,
    const float* __restrict__ bias, const float* __restrict__ g_ln,
    const float* __restrict__ b_ln, const float* __restrict__ vbias,
    const unsigned short* __restrict__ Bp, unsigned short* __restrict__ y,
    int ntiles) {
    unsigned short* Wl = (unsigned short*)smem_raw;          // 131072 B
    {
        const v4u* s = (const v4u*)Bp;
        v4u* d = (v4u*)Wl;
        for (int i = threadIdx.x; i < 8192; i += blockDim.x) d[i] = s[i];
    }
    __syncthreads();

    const int nw   = blockDim.x >> 5;
    const int wave = threadIdx.x >> 5;
    const int lane = threadIdx.x & 31;
    const int m    = lane & 15;
    const int half = lane >> 4;
    const int off0 = half << 3;
    const v8f zero = {0.f, 0.f, 0.f, 0.f, 0.f, 0.f, 0.f, 0.f};

    for (int tile = blockIdx.x * nw + wave; tile < ntiles; tile += gridDim.x * nw) {
        const int rowbase = tile << 4;
        int arow_i = rowbase + m;
        if (arow_i >= NTOT) arow_i = NTOT - 1;
        const float* arow = (arow_i >= NND) ? (vemb + (size_t)(arow_i - NND) * 256)
                                            : (x + (size_t)arow_i * 256);

        v8f acc[16];
#pragma unroll
        for (int nt = 0; nt < 16; ++nt) acc[nt] = zero;

        for (int kt = 0; kt < 8; ++kt) {
            Frag16 a;
            const float* ap = arow + kt * 32 + off0;
#pragma unroll
            for (int j = 0; j < 8; ++j) {
                a.u[j]     = f2bf(ap[j]);
                a.u[8 + j] = f2bf(ap[16 + j]);
            }
#pragma unroll
            for (int nt = 0; nt < 16; ++nt) {
                Frag16 b = load_bfrag(Wl, kt * 16 + nt, lane);   // ds_load_b128 x2
                acc[nt] = wmma_bf16(a.v, b.v, acc[nt]);
            }
        }

        // Epilogue: +bias, LayerNorm over 256 cols, ELU, vnode bias, store bf16
        const int col = m;                 // C layout: lane&15 = column in 16-tile
        v8f s = zero, s2 = zero;
#pragma unroll
        for (int nt = 0; nt < 16; ++nt) {
            const float bb = bias[nt * 16 + col];
#pragma unroll
            for (int si = 0; si < 8; ++si) {
                float t = acc[nt][si] + bb;
                acc[nt][si] = t;
                s[si]  += t;
                s2[si] += t * t;
            }
        }
#pragma unroll
        for (int mask = 1; mask <= 8; mask <<= 1) {
#pragma unroll
            for (int si = 0; si < 8; ++si) {
                s[si]  += __shfl_xor(s[si],  mask, 32);
                s2[si] += __shfl_xor(s2[si], mask, 32);
            }
        }
        float mean[8], rstd[8];
#pragma unroll
        for (int si = 0; si < 8; ++si) {
            mean[si] = s[si] * (1.0f / 256.0f);
            float var = s2[si] * (1.0f / 256.0f) - mean[si] * mean[si];
            rstd[si] = rsqrtf(var + 1e-5f);
        }
#pragma unroll
        for (int nt = 0; nt < 16; ++nt) {
            const int n = nt * 16 + col;
            const float g = g_ln[n], bl = b_ln[n];
#pragma unroll
            for (int si = 0; si < 8; ++si) {
                const int r = rowbase + si + (half << 3);
                float v = (acc[nt][si] - mean[si]) * rstd[si] * g + bl;
                v = elu(v);
                if (r >= NND && r < NTOT) v += vbias[(size_t)(r - NND) * 256 + n];
                if (r < NTOT) y[(size_t)r * 256 + n] = f2bf(v);
            }
        }
    }
}

// ---------------------------------------------------------------------------
// Layer 2: h2 = elu(h1 @ W2 + b) ; rows >= N get +vnode_bias_dcd.  bf16 in/out.
// Persistent blocks; packed W2 (128 KB) staged in LDS once per block.
// A and B fragments software-pipelined 1 deep (fits comfortably in 256 VGPRs:
// 128 acc + 4x8 fragment regs + addresses).
// ---------------------------------------------------------------------------
__global__ __launch_bounds__(256)
__attribute__((amdgpu_waves_per_eu(4)))
void layer2_kernel(
    const unsigned short* __restrict__ h, const float* __restrict__ bias,
    const float* __restrict__ vbias, const unsigned short* __restrict__ Bp,
    unsigned short* __restrict__ y, int ntiles) {
    unsigned short* Wl = (unsigned short*)smem_raw;          // 131072 B
    {
        const v4u* s = (const v4u*)Bp;
        v4u* d = (v4u*)Wl;
        for (int i = threadIdx.x; i < 8192; i += blockDim.x) d[i] = s[i];
    }
    __syncthreads();

    const int nw   = blockDim.x >> 5;
    const int wave = threadIdx.x >> 5;
    const int lane = threadIdx.x & 31;
    const int m    = lane & 15;
    const int half = lane >> 4;
    const int off0 = half << 3;
    const v8f zero = {0.f, 0.f, 0.f, 0.f, 0.f, 0.f, 0.f, 0.f};

    for (int tile = blockIdx.x * nw + wave; tile < ntiles; tile += gridDim.x * nw) {
        const int rowbase = tile << 4;
        int arow_i = rowbase + m;
        if (arow_i >= NTOT) arow_i = NTOT - 1;
        const unsigned short* arow = h + (size_t)arow_i * 256;

        v8f acc[16];
#pragma unroll
        for (int nt = 0; nt < 16; ++nt) acc[nt] = zero;

        Frag16 b = load_bfrag(Wl, 0, lane);
        Frag16 a = load_afrag16(arow, off0);
        for (int kt = 0; kt < 8; ++kt) {
            Frag16 an = (kt < 7) ? load_afrag16(arow + (kt + 1) * 32, off0) : a;
#pragma unroll
            for (int nt = 0; nt < 16; ++nt) {
                const int nf = kt * 16 + nt + 1;
                Frag16 bn = load_bfrag(Wl, nf < 128 ? nf : 0, lane);
                acc[nt] = wmma_bf16(a.v, b.v, acc[nt]);
                b = bn;
            }
            a = an;
        }

        const int col = m;
#pragma unroll
        for (int nt = 0; nt < 16; ++nt) {
            const int n = nt * 16 + col;
            const float bb = bias[n];
#pragma unroll
            for (int si = 0; si < 8; ++si) {
                const int r = rowbase + si + (half << 3);
                float v = elu(acc[nt][si] + bb);
                if (r >= NND && r < NTOT) v += vbias[(size_t)(r - NND) * 256 + n];
                if (r < NTOT) y[(size_t)r * 256 + n] = f2bf(v);
            }
        }
    }
}

// ---------------------------------------------------------------------------
// Aggregate + output head (fused):
//   nx = elu(LN(concat(h2[row], h2[N+map[row]]) @ Wa + b_ag))
//   out[row] = nx @ Wo + b_out
// LDS: packed Wa (256 KB) staged once per block + 64 KB wave-private transpose
// slices = 320 KB (full WGP LDS, one workgroup per WGP).  Wo (32 KB) from L2.
// Only 2 waves/SIMD resident here, so the 1-deep A/B fragment pipelining is
// the main latency-hiding mechanism in this kernel.
// ---------------------------------------------------------------------------
__global__ __launch_bounds__(256)
__attribute__((amdgpu_waves_per_eu(4)))
void aggr_out_kernel(
    const unsigned short* __restrict__ h2, const int* __restrict__ mapping,
    const float* __restrict__ b_ag, const float* __restrict__ g_ln,
    const float* __restrict__ b_ln, const unsigned short* __restrict__ Bpa,
    const unsigned short* __restrict__ Bpo, const float* __restrict__ b_out,
    float* __restrict__ out, int ntiles) {
    unsigned short* WaL = (unsigned short*)smem_raw;              // 262144 B
    unsigned short* nxL = (unsigned short*)(smem_raw + 262144);   // 65536 B
    {
        const v4u* s = (const v4u*)Bpa;
        v4u* d = (v4u*)WaL;
        for (int i = threadIdx.x; i < 16384; i += blockDim.x) d[i] = s[i];
    }
    __syncthreads();

    const int nw   = blockDim.x >> 5;
    const int wave = threadIdx.x >> 5;
    const int lane = threadIdx.x & 31;
    const int m    = lane & 15;
    const int half = lane >> 4;
    const int off0 = half << 3;
    const v8f zero = {0.f, 0.f, 0.f, 0.f, 0.f, 0.f, 0.f, 0.f};
    unsigned short* myl = nxL + wave * 4096;      // 16 x 256 bf16 slice

    for (int tile = blockIdx.x * nw + wave; tile < ntiles; tile += gridDim.x * nw) {
        const int rowbase = tile << 4;
        const int row = rowbase + m;              // always < NND (N % 16 == 0)
        const unsigned short* a0 = h2 + (size_t)row * 256;
        const unsigned short* a1 = h2 + (size_t)(NND + mapping[row]) * 256;

        v8f acc[16];
#pragma unroll
        for (int nt = 0; nt < 16; ++nt) acc[nt] = zero;

        Frag16 b = load_bfrag(WaL, 0, lane);
        Frag16 a = load_afrag16(a0, off0);
        for (int kt = 0; kt < 16; ++kt) {         // K = 512
            const int ktn = kt + 1;
            const unsigned short* arn = (ktn < 8)  ? (a0 + ktn * 32)
                                      : (ktn < 16) ? (a1 + (ktn - 8) * 32)
                                                   : a0;
            Frag16 an = load_afrag16(arn, off0);
#pragma unroll
            for (int nt = 0; nt < 16; ++nt) {
                const int nf = kt * 16 + nt + 1;
                Frag16 bn = load_bfrag(WaL, nf < 256 ? nf : 0, lane);
                acc[nt] = wmma_bf16(a.v, b.v, acc[nt]);
                b = bn;
            }
            a = an;
        }

        // LN + ELU epilogue -> wave-private LDS slice (row-major)
        const int col = m;
        v8f s = zero, s2 = zero;
#pragma unroll
        for (int nt = 0; nt < 16; ++nt) {
            const float bb = b_ag[nt * 16 + col];
#pragma unroll
            for (int si = 0; si < 8; ++si) {
                float t = acc[nt][si] + bb;
                acc[nt][si] = t;
                s[si]  += t;
                s2[si] += t * t;
            }
        }
#pragma unroll
        for (int mask = 1; mask <= 8; mask <<= 1) {
#pragma unroll
            for (int si = 0; si < 8; ++si) {
                s[si]  += __shfl_xor(s[si],  mask, 32);
                s2[si] += __shfl_xor(s2[si], mask, 32);
            }
        }
        float mean[8], rstd[8];
#pragma unroll
        for (int si = 0; si < 8; ++si) {
            mean[si] = s[si] * (1.0f / 256.0f);
            float var = s2[si] * (1.0f / 256.0f) - mean[si] * mean[si];
            rstd[si] = rsqrtf(var + 1e-5f);
        }
#pragma unroll
        for (int nt = 0; nt < 16; ++nt) {
            const int n = nt * 16 + col;
            const float g = g_ln[n], bl = b_ln[n];
#pragma unroll
            for (int si = 0; si < 8; ++si) {
                float v = (acc[nt][si] - mean[si]) * rstd[si] * g + bl;
                myl[(si + (half << 3)) * 256 + n] = f2bf(elu(v));
            }
        }

        // Second GEMM: nx(16x256) @ Wo(256x64) + b_out -> f32 output
        // (DS ops are in-order per wave; slice is wave-private, no barrier.)
        const unsigned short* arow2 = myl + m * 256;
        v8f acc2[4];
#pragma unroll
        for (int nt = 0; nt < 4; ++nt) acc2[nt] = zero;
        Frag16 bo2 = load_bfrag(Bpo, 0, lane);
        Frag16 a2 = load_afrag16(arow2, off0);
        for (int kt = 0; kt < 8; ++kt) {
            Frag16 an = (kt < 7) ? load_afrag16(arow2 + (kt + 1) * 32, off0) : a2;
#pragma unroll
            for (int nt = 0; nt < 4; ++nt) {
                const int nf = kt * 4 + nt + 1;
                Frag16 bn = load_bfrag(Bpo, nf < 32 ? nf : 0, lane);  // L2/WGP$-hot
                acc2[nt] = wmma_bf16(a2.v, bo2.v, acc2[nt]);
                bo2 = bn;
            }
            a2 = an;
        }
#pragma unroll
        for (int nt = 0; nt < 4; ++nt) {
            const int n = nt * 16 + col;
            const float bo = b_out[n];
#pragma unroll
            for (int si = 0; si < 8; ++si) {
                const int r = rowbase + si + (half << 3);
                out[(size_t)r * 64 + n] = acc2[nt][si] + bo;
            }
        }
    }
}

// ---------------------------------------------------------------------------
extern "C" void kernel_launch(void* const* d_in, const int* in_sizes, int n_in,
                              void* d_out, int out_size, void* d_ws, size_t ws_size,
                              hipStream_t stream) {
    const float* x      = (const float*)d_in[0];
    const int*   mapping= (const int*)  d_in[1];
    const float* vemb   = (const float*)d_in[2];
    const float* vb_hid = (const float*)d_in[3];
    const float* vb_dcd = (const float*)d_in[4];
    const float* W1     = (const float*)d_in[5];
    const float* b1     = (const float*)d_in[6];
    const float* g_h    = (const float*)d_in[7];
    const float* bln_h  = (const float*)d_in[8];
    const float* W2     = (const float*)d_in[9];
    const float* b2     = (const float*)d_in[10];
    const float* g_e    = (const float*)d_in[11];
    const float* bln_e  = (const float*)d_in[12];
    const float* Wa     = (const float*)d_in[13];
    const float* ba     = (const float*)d_in[14];
    const float* Wo     = (const float*)d_in[15];
    const float* bo     = (const float*)d_in[16];
    float* out = (float*)d_out;

    // Workspace layout (bytes)
    char* ws = (char*)d_ws;
    const size_t OFF_B1 = 0;                         // 128 frags * 1KB
    const size_t OFF_B2 = OFF_B1 + 128 * 1024;       // 128 frags
    const size_t OFF_BA = OFF_B2 + 128 * 1024;       // 256 frags
    const size_t OFF_BO = OFF_BA + 256 * 1024;       // 32 frags
    const size_t OFF_H1 = OFF_BO + 32 * 1024;        // NTOT*256 bf16
    const size_t OFF_H2 = OFF_H1 + (size_t)NTOT * 256 * 2;
    unsigned short* B1 = (unsigned short*)(ws + OFF_B1);
    unsigned short* B2 = (unsigned short*)(ws + OFF_B2);
    unsigned short* Ba = (unsigned short*)(ws + OFF_BA);
    unsigned short* Bo = (unsigned short*)(ws + OFF_BO);
    unsigned short* h1 = (unsigned short*)(ws + OFF_H1);
    unsigned short* h2 = (unsigned short*)(ws + OFF_H2);

    // Opt in to large dynamic LDS (WGP has 320 KB).
    (void)hipFuncSetAttribute((const void*)layer1_kernel,
                              hipFuncAttributeMaxDynamicSharedMemorySize, 131072);
    (void)hipFuncSetAttribute((const void*)layer2_kernel,
                              hipFuncAttributeMaxDynamicSharedMemorySize, 131072);
    (void)hipFuncSetAttribute((const void*)aggr_out_kernel,
                              hipFuncAttributeMaxDynamicSharedMemorySize, 327680);

    // Pack weights to bf16 WMMA-B fragments (cheap, reused by 200k rows)
    pack_kernel<<<32, 128, 0, stream>>>(W1, B1, 256, 256);
    pack_kernel<<<32, 128, 0, stream>>>(W2, B2, 256, 256);
    pack_kernel<<<64, 128, 0, stream>>>(Wa, Ba, 512, 256);
    pack_kernel<<< 8, 128, 0, stream>>>(Wo, Bo, 256, 64);

    const int tiles12 = (NTOT + 15) / 16;            // 12501
    const int tiles3  = NND / 16;                    // 12500
    const int pblocks = 512;                         // persistent blocks

    layer1_kernel<<<pblocks, 256, 131072, stream>>>(x, vemb, b1, g_h, bln_h, vb_hid,
                                                    B1, h1, tiles12);
    layer2_kernel<<<pblocks, 256, 131072, stream>>>(h1, b2, vb_dcd, B2, h2, tiles12);
    aggr_out_kernel<<<pblocks, 256, 327680, stream>>>(h2, mapping, ba, g_e, bln_e,
                                                      Ba, Bo, bo, out, tiles3);
}
